// ChannelMerger_17660905521243
// MI455X (gfx1250) — matrix operand order repaired and checked by reading, not compiled, not run on hardware
//
#include <hip/hip_runtime.h>

// ---------------------------------------------------------------------------
// ChannelMerger for MI455X (gfx1250, wave32, WMMA f32_16x16x32_f16)
//
// out[b,o,t] = sum_c softmax_c( heads[o,:] . fourier_emb(pos[b,c,:]) ) * meg[b,c,t]
//
// Dominant stage: per batch [270x273]x[273x8192]. HBM floor ~570MB => ~24.5us
// @ 23.3TB/s; f16-WMMA compute ~8us => memory bound. K3 therefore:
//  - 4 column-chunks per block (weights staged to LDS once, L2-resident)
//  - double-buffered meg tile in LDS so HBM streaming overlaps WMMA sweep
// ---------------------------------------------------------------------------

typedef __attribute__((ext_vector_type(16))) _Float16 v16h;
typedef __attribute__((ext_vector_type(8)))  _Float16 v8h;
typedef __attribute__((ext_vector_type(4)))  _Float16 v4h;
typedef __attribute__((ext_vector_type(8)))  float    v8f;
typedef __attribute__((ext_vector_type(4)))  float    v4f;

#define B_SZ   32
#define C_CH   273          // input channels (K of merge GEMM)
#define CHOUT  270          // output channels (M)
#define T_LEN  8192         // time (N)
#define PD     288          // POS_DIM (= 9*32, exact K tiling)
#define MT     17           // M tiles: 270 -> 272
#define KT     9            // K tiles of 32: 273 -> 288
#define NT1    18           // N tiles for scores GEMM: 273 -> 288
#define WSW_ELE (MT*KT*2*32*8)   // 78336 f16 per swizzled A matrix (156,672 B)
#define NBLK   128          // columns of meg per chunk in K3
#define NCHUNK 4            // chunks per block (512 columns)
#define LDSB_STRIDE 304     // f16 row stride (multiple of 16 -> 32B-aligned chunks)

// A-fragment swizzle (this code's contiguous k order, not ISA table order, 16-bit A 16x32):
//   lane<16 : VGPR0-3 = K 0..7,   VGPR4-7 = K 8..15   (row M = lane)
//   lane>=16: VGPR0-3 = K 16..23, VGPR4-7 = K 24..31  (row M = lane-16)
// stored as chunks of 8 f16: chunk = ((mt*KT+ks)*2+half)*32 + lane
__device__ __forceinline__ int swz_idx(int o, int k) {
  int mt = o >> 4, lr = o & 15;
  int ks = k >> 5, km = k & 31;
  int q = km >> 3, within = km & 7;
  int lane = lr + ((q >> 1) ? 16 : 0);
  int half = q & 1;
  return ((((mt * KT + ks) * 2 + half) * 32) + lane) * 8 + within;
}

__device__ __forceinline__ v16h load_afrag(const _Float16* base, int mt, int ks, int lane) {
  const v8h* p0 = (const v8h*)(base + (((mt * KT + ks) * 2 + 0) * 32 + lane) * 8);
  const v8h* p1 = (const v8h*)(base + (((mt * KT + ks) * 2 + 1) * 32 + lane) * 8);
  v8h lo = *p0, hi = *p1;
  return __builtin_shufflevector(lo, hi, 0,1,2,3,4,5,6,7,8,9,10,11,12,13,14,15);
}

// ---------------- K0: swizzle heads (f32 [270][288]) -> f16 A-frag layout ---
__global__ __launch_bounds__(256)
void k0_heads_swizzle(const float* __restrict__ heads, _Float16* __restrict__ out) {
  int idx = blockIdx.x * 256 + threadIdx.x;           // covers 272*288
  int o = idx / PD, k = idx % PD;
  float v = (o < CHOUT) ? heads[o * PD + k] : 0.f;
  out[swz_idx(o, k)] = (_Float16)v;
}

// ---------------- K1: Fourier emb (into LDS, B^T layout) + scores GEMM ------
__global__ __launch_bounds__(256)
void k1_emb_scores(const float* __restrict__ positions,
                   const _Float16* __restrict__ headsSw,
                   float* __restrict__ scores) {
  extern __shared__ _Float16 ldsE[];                  // [288 ch][288 k] f16
  const int b = blockIdx.x;
  const int tid = threadIdx.x;
  const float PW = 2.f * 3.14159265358979323846f / 1.4f;   // 2*pi/width

  // emb[c,k] = cos/sin( PW*(x*i + y*j) ), k = i*12+j (cos) / 144+i*12+j (sin)
  for (int idx = tid; idx < PD * 144; idx += 256) {
    int c = idx / 144, f = idx % 144;
    float cv = 0.f, sv = 0.f;
    if (c < C_CH) {
      float x = positions[((size_t)b * C_CH + c) * 2 + 0] + 0.2f;
      float y = positions[((size_t)b * C_CH + c) * 2 + 1] + 0.2f;
      int i = f / 12, j = f % 12;
      float loc = PW * (x * (float)i + y * (float)j);
      __sincosf(loc, &sv, &cv);
    }
    ldsE[c * PD + f]       = (_Float16)cv;
    ldsE[c * PD + 144 + f] = (_Float16)sv;
  }
  __syncthreads();

  const int w = tid >> 5, lane = tid & 31;
  const int ln = lane & 15;
  const bool hi = lane >= 16;

  for (int t = w; t < MT * NT1; t += 8) {
    int mt = t / NT1, nt = t % NT1;
    v8f acc = {};
#pragma unroll
    for (int ks = 0; ks < KT; ++ks) {
      v16h a = load_afrag(headsSw, mt, ks, lane);
      // B-frag (K=32 x N=16): lane n holds column n, 16 consecutive K in f16
      int n = nt * 16 + ln;
      int koff = ks * 32 + (hi ? 16 : 0);
      v16h bfr = *(const v16h*)(ldsE + n * PD + koff);
      acc = __builtin_amdgcn_wmma_f32_16x16x32_f16(false, a, false, bfr,
                                                   (short)0, acc, false, false);
    }
    int cIdx = nt * 16 + ln;
    if (cIdx < C_CH) {
#pragma unroll
      for (int r = 0; r < 8; ++r) {
        int o = mt * 16 + r + (hi ? 8 : 0);
        if (o < CHOUT)
          scores[((size_t)b * CHOUT + o) * C_CH + cIdx] = acc[r];
      }
    }
  }
}

// ---------------- K2: softmax over C + swizzle weights to f16 A layout ------
__global__ __launch_bounds__(256)
void k2_softmax_swizzle(const float* __restrict__ scores,
                        _Float16* __restrict__ wsw) {
  const int w = threadIdx.x >> 5, lane = threadIdx.x & 31;
  const int row = blockIdx.x * 8 + w;                 // 0 .. 32*272-1
  const int b = row / 272, o = row % 272;
  _Float16* out = wsw + (size_t)b * WSW_ELE;
  const float NEG = -__builtin_inff();

  float s[KT];
  if (o < CHOUT) {
    float m = NEG;
#pragma unroll
    for (int it = 0; it < KT; ++it) {
      int c = it * 32 + lane;
      s[it] = (c < C_CH) ? scores[((size_t)b * CHOUT + o) * C_CH + c] : NEG;
      m = fmaxf(m, s[it]);
    }
#pragma unroll
    for (int off = 16; off > 0; off >>= 1) m = fmaxf(m, __shfl_xor(m, off, 32));
    float sum = 0.f;
#pragma unroll
    for (int it = 0; it < KT; ++it) { s[it] = __expf(s[it] - m); sum += s[it]; }
#pragma unroll
    for (int off = 16; off > 0; off >>= 1) sum += __shfl_xor(sum, off, 32);
    float inv = 1.f / sum;
#pragma unroll
    for (int it = 0; it < KT; ++it) s[it] *= inv;
  } else {
#pragma unroll
    for (int it = 0; it < KT; ++it) s[it] = 0.f;
  }
#pragma unroll
  for (int it = 0; it < KT; ++it) {
    int c = it * 32 + lane;                           // 0..287 exactly
    float v = (c < C_CH) ? s[it] : 0.f;
    out[swz_idx(o, c)] = (_Float16)v;
  }
}

// Fill one transposed f16 meg tile [col][k] into LDS; 4 k-values per thread
// per iteration -> packed ds_store_b64. Global loads per-row coalesced.
__device__ __forceinline__ void fill_btile(_Float16* __restrict__ dst,
                                           const float* __restrict__ megB,
                                           int tid) {
  for (int e = tid; e < NBLK * (PD / 4); e += 256) {
    int n = e & (NBLK - 1);
    int kg = e >> 7;                 // 0..71
    int k = kg * 4;
    v4h p;
#pragma unroll
    for (int u = 0; u < 4; ++u) {
      int kk = k + u;
      float v = (kk < C_CH) ? megB[(size_t)kk * T_LEN + n] : 0.f;
      p[u] = (_Float16)v;
    }
    *(v4h*)(dst + n * LDSB_STRIDE + k) = p;
  }
}

// ---------------- K3: merge GEMM  out[b] = W[b](270x273) x meg[b](273x8192) -
__global__ __launch_bounds__(256, 1)
void k3_merge(const float* __restrict__ meg, const _Float16* __restrict__ wsw,
              float* __restrict__ out) {
  extern __shared__ char smem[];
  _Float16* ldsW  = (_Float16*)smem;                          // 156,672 B
  _Float16* ldsB0 = (_Float16*)(smem + WSW_ELE * 2);          //  77,824 B
  _Float16* ldsB1 = ldsB0 + NBLK * LDSB_STRIDE;               //  77,824 B

  const int b   = blockIdx.x >> 4;                  // 32 batches
  const int nsp = blockIdx.x & 15;                  // 16 spans of 512 columns
  const int col0 = nsp * (NBLK * NCHUNK);
  const int tid = threadIdx.x;

  const float* megBase = meg + (size_t)b * C_CH * T_LEN + col0;

  // stage swizzled weights -> LDS (L2-resident; reused for 4 chunks)
  {
    const v4f* src = (const v4f*)(wsw + (size_t)b * WSW_ELE);
    v4f* dst = (v4f*)ldsW;
    for (int i = tid; i < WSW_ELE / 8; i += 256) dst[i] = src[i];
  }
  // stage first meg chunk
  fill_btile(ldsB0, megBase, tid);
  __syncthreads();

  const int w = tid >> 5, lane = tid & 31;
  const int ln = lane & 15;
  const bool hi = lane >= 16;

  for (int ic = 0; ic < NCHUNK; ++ic) {
    _Float16* bufCur = (ic & 1) ? ldsB1 : ldsB0;
    _Float16* bufNxt = (ic & 1) ? ldsB0 : ldsB1;

    // cache this wave's 9 B-fragments (72 VGPRs); bufCur then free
    v16h bf[KT];
    {
      const _Float16* lb = bufCur + (w * 16 + ln) * LDSB_STRIDE;
#pragma unroll
      for (int ks = 0; ks < KT; ++ks)
        bf[ks] = *(const v16h*)(lb + ks * 32 + (hi ? 16 : 0));
    }
    // prefetch next chunk into the other buffer; overlaps the WMMA sweep
    if (ic + 1 < NCHUNK)
      fill_btile(bufNxt, megBase + (ic + 1) * NBLK, tid);

    float* outB = out + (size_t)b * CHOUT * T_LEN + col0 + ic * NBLK + w * 16 + ln;
    for (int mt = 0; mt < MT; ++mt) {
      v8f acc = {};
#pragma unroll
      for (int ks = 0; ks < KT; ++ks) {
        v16h a = load_afrag(ldsW, mt, ks, lane);
        acc = __builtin_amdgcn_wmma_f32_16x16x32_f16(false, a, false, bf[ks],
                                                     (short)0, acc, false, false);
      }
      int obase = mt * 16 + (hi ? 8 : 0);
#pragma unroll
      for (int r = 0; r < 8; ++r) {
        int o = obase + r;
        if (o < CHOUT) outB[(size_t)o * T_LEN] = acc[r];
      }
    }
    __syncthreads();   // fill of next buffer complete + this buffer reusable
  }
}

// ---------------------------------------------------------------------------
extern "C" void kernel_launch(void* const* d_in, const int* in_sizes, int n_in,
                              void* d_out, int out_size, void* d_ws, size_t ws_size,
                              hipStream_t stream) {
  const float* meg       = (const float*)d_in[0];   // [32,273,8192] f32
  const float* positions = (const float*)d_in[1];   // [32,273,2]    f32
  const float* heads     = (const float*)d_in[2];   // [270,288]     f32
  float* out = (float*)d_out;                       // [32,270,8192] f32

  char* ws = (char*)d_ws;
  const size_t HEADS_BYTES  = (size_t)WSW_ELE * 2;                 // 156,672
  const size_t SCORES_BYTES = (size_t)B_SZ * CHOUT * C_CH * 4;     // 9,434,880
  _Float16* wsHeads  = (_Float16*)(ws);
  float*    wsScores = (float*)(ws + HEADS_BYTES);
  _Float16* wsW      = (_Float16*)(ws + HEADS_BYTES + SCORES_BYTES);

  // K0: 272*288 elements
  k0_heads_swizzle<<<(272 * PD) / 256, 256, 0, stream>>>(heads, wsHeads);
  // K1: one block per batch; LDS = 288*288 f16
  k1_emb_scores<<<B_SZ, 256, PD * PD * 2, stream>>>(positions, wsHeads, wsScores);
  // K2: 8 rows per block over 32*272 padded rows
  k2_softmax_swizzle<<<(B_SZ * 272) / 8, 256, 0, stream>>>(wsScores, wsW);
  // K3: 32 batches x 16 spans; LDS = weights + double-buffered B tile
  const int k3_lds = WSW_ELE * 2 + 2 * NBLK * LDSB_STRIDE * 2;     // 312,320 B
  k3_merge<<<B_SZ * 16, 256, k3_lds, stream>>>(meg, wsW, out);
}
